// Attention_61014305407316
// MI455X (gfx1250) — matrix operand order repaired and checked by reading, not compiled, not run on hardware
//
#include <hip/hip_runtime.h>
#include <stdint.h>

#define DIM    512
#define HEADS  8
#define DHEAD  64
#define INNER  512
#define BATCH  4
#define SEQ    1024
#define BH     (BATCH*HEADS)      // 32
#define KKEEP  716                // int(1024*0.7)
#define THRIDX (SEQ - KKEEP)      // 308: smallest kept value (ascending sort)
#define QSCALE 0.125f             // 64^-0.5

typedef __attribute__((ext_vector_type(16))) __bf16 v16bf;
typedef __attribute__((ext_vector_type(8)))  __bf16 v8bf;
typedef __attribute__((ext_vector_type(8)))  float  v8f;

// ---------- bf16 helpers (round-to-nearest-even), ushort storage ----------
__device__ __forceinline__ unsigned short f2bfbits(float f) {
  union { float f; unsigned u; } x; x.f = f;
  unsigned u = x.u;
  unsigned r = u + 0x7FFFu + ((u >> 16) & 1u);
  return (unsigned short)(r >> 16);
}

union fragu { v16bf v; v8bf h[2]; };

// A (16x32 MxK) fragment, row-major A, k order chosen by this code (not the ISA table order):
//   elems 0..15 -> K = k0 + 16*half + 0..15 (contiguous, matches B)
__device__ __forceinline__ v16bf load_a_frag(const unsigned short* base, size_t ld,
                                             int row, int k0, int half) {
  const v8bf* p = (const v8bf*)(base + (size_t)row * ld + k0);
  fragu f;
  f.h[0] = p[2 * half];
  f.h[1] = p[2 * half + 1];
  return f.v;
}
// B (32x16 KxN) fragment from B^T stored row-major ([N,K]):
//   elems 0..15 -> K = k0 + 16*half + 0..15 (contiguous), col fixed per lane
__device__ __forceinline__ v16bf load_b_frag(const unsigned short* baseT, size_t ld,
                                             int col, int k0, int half) {
  return *(const v16bf*)(baseT + (size_t)col * ld + k0 + 16 * half);
}

// =====================================================================
// K0a: x (fp32 [4096,512]) -> bf16 row-major
// =====================================================================
__global__ void __launch_bounds__(256) cvt_x_kernel(const float* __restrict__ in,
                                                    unsigned short* __restrict__ out) {
  const size_t i = (size_t)blockIdx.x * blockDim.x + threadIdx.x;
  out[i] = f2bfbits(in[i]);
}

// =====================================================================
// K0b: transpose + convert: in [R,C] fp32 row-major -> out [C,R] bf16
// =====================================================================
__global__ void __launch_bounds__(256) transpose_cvt_kernel(const float* __restrict__ in,
                                                            unsigned short* __restrict__ outT,
                                                            int R, int C) {
  const size_t i = (size_t)blockIdx.x * blockDim.x + threadIdx.x;   // exact grid
  const int r = (int)(i / C), c = (int)(i % C);
  outT[(size_t)c * R + r] = f2bfbits(in[i]);
}

// =====================================================================
// K1: qkv = x @ w_qkv (A = xbf row-major, B = wqkvT [1536,512]).
//     Scatter: q*scale -> [B,H,N,D]; k -> [B,H,N,D] (row-major!);
//     v -> transposed [B,H,D,N].  Wave computes a 16x64 tile.
// =====================================================================
__global__ void __launch_bounds__(256) qkv_kernel(
    const unsigned short* __restrict__ xbf, const unsigned short* __restrict__ wqkvT,
    unsigned short* __restrict__ qbf, unsigned short* __restrict__ kbf,
    unsigned short* __restrict__ vtbf) {
  const int gw   = (blockIdx.x * blockDim.x + threadIdx.x) >> 5;
  const int lane = threadIdx.x & 31;
  const int half = lane >> 4;
  const int l15  = lane & 15;
  const int NSTRIP = (3 * INNER) / 64;   // 24
  const int mt = gw / NSTRIP;            // 0..255
  const int ns = gw % NSTRIP;
  const int m0 = mt * 16, n0 = ns * 64;

  v8f acc[4];
  #pragma unroll
  for (int t = 0; t < 4; ++t)
    #pragma unroll
    for (int r = 0; r < 8; ++r) acc[t][r] = 0.0f;

  for (int k0 = 0; k0 < DIM; k0 += 32) {
    const v16bf a = load_a_frag(xbf, DIM, m0 + l15, k0, half);
    #pragma unroll
    for (int t = 0; t < 4; ++t) {
      const v16bf b = load_b_frag(wqkvT, DIM, n0 + t * 16 + l15, k0, half);
      acc[t] = __builtin_amdgcn_wmma_f32_16x16x32_bf16(
                   false, a, false, b, (short)0, acc[t], false, false);
    }
  }

  #pragma unroll
  for (int t = 0; t < 4; ++t) {
    const int colb = n0 + t * 16;          // wave-uniform
    const int seg  = colb >> 9;            // 0=q, 1=k, 2=v (uniform)
    const int cb   = colb - seg * 512;
    const int h    = cb >> 6;              // head (uniform: cb%64 in {0,16,32,48})
    const int d    = (cb & 63) + l15;
    #pragma unroll
    for (int r = 0; r < 8; ++r) {
      const int row = m0 + r + 8 * half;   // 0..4095
      const int bb  = row >> 10, nn = row & (SEQ - 1);
      const float val = acc[t][r];
      const size_t bhh = (size_t)(bb * HEADS + h);
      if (seg == 0) {
        qbf[(bhh * SEQ + nn) * DHEAD + d] = f2bfbits(val * QSCALE);
      } else if (seg == 1) {
        kbf[(bhh * SEQ + nn) * DHEAD + d] = f2bfbits(val);
      } else {
        vtbf[(bhh * DHEAD + d) * SEQ + nn] = f2bfbits(val);
      }
    }
  }
}

// =====================================================================
// K2: dots[bh] = q[bh] @ k[bh]^T  (both [1024,64] row-major; k row-major
//     IS (k^T)^T, so B-fragments are contiguous). fp32 out.
// =====================================================================
__global__ void __launch_bounds__(256) dots_kernel(
    const unsigned short* __restrict__ qbf, const unsigned short* __restrict__ kbf,
    float* __restrict__ dots) {
  const int gw   = (blockIdx.x * blockDim.x + threadIdx.x) >> 5;
  const int lane = threadIdx.x & 31;
  const int half = lane >> 4;
  const int l15  = lane & 15;
  const int bh = gw >> 10;
  const int u  = gw & 1023;
  const int mt = u >> 4;        // 64 M-tiles
  const int ns = u & 15;        // 16 N-strips of 64
  const int m0 = mt * 16, n0 = ns * 64;
  const unsigned short* q = qbf + (size_t)bh * SEQ * DHEAD;
  const unsigned short* k = kbf + (size_t)bh * SEQ * DHEAD;

  v8f acc[4];
  #pragma unroll
  for (int t = 0; t < 4; ++t)
    #pragma unroll
    for (int r = 0; r < 8; ++r) acc[t][r] = 0.0f;

  #pragma unroll
  for (int k0 = 0; k0 < DHEAD; k0 += 32) {
    const v16bf a = load_a_frag(q, DHEAD, m0 + l15, k0, half);
    #pragma unroll
    for (int t = 0; t < 4; ++t) {
      const v16bf b = load_b_frag(k, DHEAD, n0 + t * 16 + l15, k0, half);
      acc[t] = __builtin_amdgcn_wmma_f32_16x16x32_bf16(
                   false, a, false, b, (short)0, acc[t], false, false);
    }
  }

  float* dbase = dots + (size_t)bh * SEQ * SEQ;
  #pragma unroll
  for (int t = 0; t < 4; ++t)
    #pragma unroll
    for (int r = 0; r < 8; ++r)
      dbase[(size_t)(m0 + r + 8 * half) * SEQ + n0 + t * 16 + l15] = acc[t][r];
}

// =====================================================================
// K3: per-row top-k threshold (bitonic sort in LDS) + masked softmax.
//     Writes bf16 probabilities IN PLACE over the first half of each
//     fp32 dots row (safe: row fully read before write; rows disjoint).
// =====================================================================
__global__ void __launch_bounds__(512) topk_softmax_kernel(float* __restrict__ dots) {
  __shared__ float ssort[SEQ];
  __shared__ float red[512];
  const int row = blockIdx.x;               // 0..BH*SEQ-1
  const int tid = threadIdx.x;              // 0..511
  float* drow = dots + (size_t)row * SEQ;
  const float v0 = drow[tid];
  const float v1 = drow[tid + 512];
  ssort[tid] = v0; ssort[tid + 512] = v1;
  __syncthreads();

  // bitonic sort, ascending
  for (int k = 2; k <= SEQ; k <<= 1) {
    for (int j = k >> 1; j > 0; j >>= 1) {
      const int l = ((tid / j) * 2 * j) + (tid % j);
      const int p = l + j;
      const bool asc = ((l & k) == 0);
      const float a = ssort[l], b = ssort[p];
      if ((a > b) == asc) { ssort[l] = b; ssort[p] = a; }
      __syncthreads();
    }
  }

  const float thr = ssort[THRIDX];   // smallest kept value (top-716)
  const float mx  = ssort[SEQ - 1];

  const float p0 = (v0 >= thr) ? __expf(v0 - mx) : 0.0f;
  const float p1 = (v1 >= thr) ? __expf(v1 - mx) : 0.0f;
  red[tid] = p0 + p1;
  __syncthreads();
  for (int s = 256; s > 0; s >>= 1) {
    if (tid < s) red[tid] += red[tid + s];
    __syncthreads();
  }
  const float inv = 1.0f / red[0];

  unsigned short* arow = reinterpret_cast<unsigned short*>(dots) + (size_t)row * (2 * SEQ);
  arow[tid]       = f2bfbits(p0 * inv);
  arow[tid + 512] = f2bfbits(p1 * inv);
}

// =====================================================================
// K4: o[bh] = attn[bh] @ v[bh]  (A = attn rows in aliased dots buffer,
//     B from vT [B,H,D,N] so fragments are contiguous). Writes merged
//     heads into ao (bf16 [B,N,INNER]). Wave: 16x64 (full D).
// =====================================================================
__global__ void __launch_bounds__(256) av_kernel(
    const unsigned short* __restrict__ attn,   // aliased dots buffer, row stride 2*SEQ
    const unsigned short* __restrict__ vtbf,
    unsigned short* __restrict__ aobf) {
  const int gw   = (blockIdx.x * blockDim.x + threadIdx.x) >> 5;
  const int lane = threadIdx.x & 31;
  const int half = lane >> 4;
  const int l15  = lane & 15;
  const int bh = gw >> 6;
  const int mt = gw & 63;
  const int m0 = mt * 16;
  const unsigned short* vt = vtbf + (size_t)bh * DHEAD * SEQ;

  v8f acc[4];
  #pragma unroll
  for (int t = 0; t < 4; ++t)
    #pragma unroll
    for (int r = 0; r < 8; ++r) acc[t][r] = 0.0f;

  for (int k0 = 0; k0 < SEQ; k0 += 32) {
    const v16bf a = load_a_frag(attn, 2 * SEQ, bh * SEQ + m0 + l15, k0, half);
    #pragma unroll
    for (int t = 0; t < 4; ++t) {
      const v16bf b = load_b_frag(vt, SEQ, t * 16 + l15, k0, half);
      acc[t] = __builtin_amdgcn_wmma_f32_16x16x32_bf16(
                   false, a, false, b, (short)0, acc[t], false, false);
    }
  }

  const int b = bh >> 3, h = bh & 7;
  #pragma unroll
  for (int t = 0; t < 4; ++t)
    #pragma unroll
    for (int r = 0; r < 8; ++r) {
      const int m = m0 + r + 8 * half;
      const int d = t * 16 + l15;
      aobf[((size_t)(b * SEQ + m)) * INNER + h * DHEAD + d] = f2bfbits(acc[t][r]);
    }
}

// =====================================================================
// K5: out = ao @ w_out + b_out  (B from woutT [512,512] bf16)
// =====================================================================
__global__ void __launch_bounds__(256) out_kernel(
    const unsigned short* __restrict__ aobf, const unsigned short* __restrict__ woutT,
    const float* __restrict__ bout, float* __restrict__ out) {
  const int gw   = (blockIdx.x * blockDim.x + threadIdx.x) >> 5;
  const int lane = threadIdx.x & 31;
  const int half = lane >> 4;
  const int l15  = lane & 15;
  const int NS = DIM / 64;               // 8
  const int mt = gw / NS, ns = gw % NS;
  const int m0 = mt * 16, n0 = ns * 64;

  v8f acc[4];
  #pragma unroll
  for (int t = 0; t < 4; ++t)
    #pragma unroll
    for (int r = 0; r < 8; ++r) acc[t][r] = 0.0f;

  for (int k0 = 0; k0 < INNER; k0 += 32) {
    const v16bf a = load_a_frag(aobf, INNER, m0 + l15, k0, half);
    #pragma unroll
    for (int t = 0; t < 4; ++t) {
      const v16bf b = load_b_frag(woutT, INNER, n0 + t * 16 + l15, k0, half);
      acc[t] = __builtin_amdgcn_wmma_f32_16x16x32_bf16(
                   false, a, false, b, (short)0, acc[t], false, false);
    }
  }

  #pragma unroll
  for (int t = 0; t < 4; ++t) {
    const int col = n0 + t * 16 + l15;
    const float bias = bout[col];
    #pragma unroll
    for (int r = 0; r < 8; ++r) {
      const int row = m0 + r + 8 * half;
      out[(size_t)row * DIM + col] = acc[t][r] + bias;
    }
  }
}

// =====================================================================
extern "C" void kernel_launch(void* const* d_in, const int* in_sizes, int n_in,
                              void* d_out, int out_size, void* d_ws, size_t ws_size,
                              hipStream_t stream) {
  const float* x    = (const float*)d_in[0];
  const float* wqkv = (const float*)d_in[1];
  const float* wout = (const float*)d_in[2];
  const float* bout = (const float*)d_in[3];
  float* out = (float*)d_out;

  char* ws = (char*)d_ws;
  float* dots = (float*)ws;                                  // 32*1024*1024 f32 = 128 MB
  size_t off = (size_t)BH * SEQ * SEQ * sizeof(float);
  unsigned short* qbf   = (unsigned short*)(ws + off); off += (size_t)BH * SEQ * DHEAD * 2;
  unsigned short* kbf   = (unsigned short*)(ws + off); off += (size_t)BH * SEQ * DHEAD * 2;
  unsigned short* vtbf  = (unsigned short*)(ws + off); off += (size_t)BH * SEQ * DHEAD * 2;
  unsigned short* aobf  = (unsigned short*)(ws + off); off += (size_t)BATCH * SEQ * INNER * 2;
  unsigned short* xbf   = (unsigned short*)(ws + off); off += (size_t)BATCH * SEQ * DIM * 2;
  unsigned short* wqkvT = (unsigned short*)(ws + off); off += (size_t)DIM * 3 * INNER * 2;
  unsigned short* woutT = (unsigned short*)(ws + off);

  // --- precision staging: bf16 convert x, transpose+convert weights ---
  cvt_x_kernel<<<(BATCH * SEQ * DIM) / 256, 256, 0, stream>>>(x, xbf);
  transpose_cvt_kernel<<<(DIM * 3 * INNER) / 256, 256, 0, stream>>>(wqkv, wqkvT, DIM, 3 * INNER);
  transpose_cvt_kernel<<<(INNER * DIM) / 256, 256, 0, stream>>>(wout, woutT, INNER, DIM);

  // waves: 6144 = 256 M-tiles * 24 N-strips
  qkv_kernel<<<768, 256, 0, stream>>>(xbf, wqkvT, qbf, kbf, vtbf);
  // waves: 32768 = 32 bh * 64 M-tiles * 16 N-strips
  dots_kernel<<<4096, 256, 0, stream>>>(qbf, kbf, dots);
  // one 512-thread block per row
  topk_softmax_kernel<<<BH * SEQ, 512, 0, stream>>>(dots);
  // waves: 2048 = 32 bh * 64 M-tiles
  av_kernel<<<256, 256, 0, stream>>>((const unsigned short*)dots, vtbf, aobf);
  // waves: 2048 = 256 M-tiles * 8 N-strips
  out_kernel<<<256, 256, 0, stream>>>(aobf, woutT, bout, out);
}
